// MixtureOfExperts_13048110645952
// MI455X (gfx1250) — matrix operand order repaired and checked by reading, not compiled, not run on hardware
//
#include <hip/hip_runtime.h>
#include <hip/hip_bf16.h>
#include <math.h>

// Problem constants (from reference): B=2, T=2048 -> 4096 tokens
#define TOKENS   4096
#define C_DIM    1024
#define F_DIM    4096
#define E_NUM    8
#define TOPK     2
#define TILE_M   16
#define OUT_ELEMS (TOKENS * C_DIM)

// workspace layout (needs ~144 MB scratch)
#define WS_COUNTS   0
#define WS_PROBSUM  64
#define WS_TOK      128
#define WS_WL       (WS_TOK + (size_t)TOKENS * E_NUM * 4)
#define WS_XBF      ((size_t)1  << 20)   // bf16 x:        8 MB
#define WS_W1P      ((size_t)16 << 20)   // packed W1:    64 MB
#define WS_W2P      ((size_t)80 << 20)   // packed W2:    64 MB  (ends at 144 MB)

typedef __attribute__((ext_vector_type(16))) __bf16        v16bf;
typedef __attribute__((ext_vector_type(8)))  float         v8f;
typedef __attribute__((ext_vector_type(8)))  unsigned short us8;
typedef __attribute__((ext_vector_type(16))) unsigned short us16;

// fp32 -> bf16 (round to nearest even)
__device__ __forceinline__ unsigned short f2bf(float f) {
    unsigned int u = __builtin_bit_cast(unsigned int, f);
    u += 0x7FFFu + ((u >> 16) & 1u);
    return (unsigned short)(u >> 16);
}

__device__ __forceinline__ v16bf concat8(us8 lo, us8 hi) {
    us16 r = __builtin_shufflevector(lo, hi, 0, 1, 2, 3, 4, 5, 6, 7,
                                             8, 9, 10, 11, 12, 13, 14, 15);
    return __builtin_bit_cast(v16bf, r);
}

// ---------------------------------------------------------------------------
// A-fragment (16 x 32 slice of a row-major bf16 LDS tile), contiguous k order (matches packed B):
// lane 0-15  row M=lane,    elems 0..15 -> K=k0..k0+15
// lane 16-31 row M=lane-16, elems 0..15 -> K=k0+16..k0+31
// +8-short row pad -> both ds_load_b128 are bank-conflict free.
// ---------------------------------------------------------------------------
__device__ __forceinline__ v16bf load_a_frag(const unsigned short* base, int stride, int k0) {
    const int lane = threadIdx.x & 31;
    const int m    = lane & 15;
    const int kb   = k0 + ((lane >> 4) << 4);
    const us8* p = (const us8*)(base + m * stride + kb);
    return concat8(p[0], p[1]);   // K = kb..kb+15
}

// ===========================================================================
// Weight repack: W[e][Kd][Nd] fp32 -> bf16 tiles in exact B-fragment order.
// Tile (tk = k/32, tn = n/16) stores 512 bf16; lane l's 16 elements are
// contiguous at [l*16 + j], with l = (n&15) | ((k>>4)&1)<<4, j = k&15.
// Hot loop then needs only 2 coalesced global_load_b128 per fragment.
// ===========================================================================
template <int Kd, int Nd>
__global__ __launch_bounds__(256) void moe_pack_w_kernel(const float* __restrict__ W,
                                                         unsigned short* __restrict__ P) {
    const long id = (long)blockIdx.x * 256 + threadIdx.x;     // over E*Kd*Nd
    const int  n  = (int)(id & (Nd - 1));
    const long r  = id >> __builtin_ctz(Nd);
    const int  k  = (int)(r & (Kd - 1));
    const int  e  = (int)(r >> __builtin_ctz(Kd));
    const int  l  = (n & 15) | (((k >> 4) & 1) << 4);
    const int  j  = k & 15;
    const long dst = ((((long)e * (Nd >> 4) + (n >> 4)) * (Kd >> 5) + (k >> 5)) * 512)
                     + l * 16 + j;
    P[dst] = f2bf(W[id]);
}

__global__ __launch_bounds__(256) void moe_pack_x_kernel(const float* __restrict__ x,
                                                         unsigned short* __restrict__ xbf) {
    const long id = (long)blockIdx.x * 256 + threadIdx.x;     // TOKENS*C_DIM
    xbf[id] = f2bf(x[id]);
}

// ===========================================================================
// Kernel 0: zero output + routing counters (harness poisons buffers with 0xAA)
// ===========================================================================
__global__ void moe_zero_kernel(float* __restrict__ out, int n_out,
                                int* __restrict__ counts, float* __restrict__ probsum) {
    long i = (long)blockIdx.x * blockDim.x + threadIdx.x;
    if (i < n_out) out[i] = 0.0f;
    if (i < E_NUM) { counts[i] = 0; probsum[i] = 0.0f; }
}

// ===========================================================================
// Kernel 1: router. One wave32 per token.
// ===========================================================================
__global__ __launch_bounds__(256) void moe_router_kernel(
    const float* __restrict__ x, const float* __restrict__ Wg,
    int* __restrict__ counts, float* __restrict__ probsum,
    int* __restrict__ tok_list, float* __restrict__ w_list) {
    const int token = blockIdx.x * 8 + (threadIdx.x >> 5);
    const int lane  = threadIdx.x & 31;
    const float* xr = x + (long)token * C_DIM;

    float acc[E_NUM];
#pragma unroll
    for (int e = 0; e < E_NUM; ++e) acc[e] = 0.0f;

    for (int i = lane; i < C_DIM; i += 32) {
        float xv = xr[i];
        const float4* wg = (const float4*)(Wg + i * E_NUM);
        float4 a = wg[0], b = wg[1];
        acc[0] += xv * a.x; acc[1] += xv * a.y; acc[2] += xv * a.z; acc[3] += xv * a.w;
        acc[4] += xv * b.x; acc[5] += xv * b.y; acc[6] += xv * b.z; acc[7] += xv * b.w;
    }
#pragma unroll
    for (int off = 16; off > 0; off >>= 1)
#pragma unroll
        for (int e = 0; e < E_NUM; ++e) acc[e] += __shfl_down(acc[e], off, 32);

    if (lane == 0) {
        float mx = acc[0];
#pragma unroll
        for (int e = 1; e < E_NUM; ++e) mx = fmaxf(mx, acc[e]);
        float p[E_NUM], s = 0.0f;
#pragma unroll
        for (int e = 0; e < E_NUM; ++e) { p[e] = __expf(acc[e] - mx); s += p[e]; }
        float inv = 1.0f / s;
#pragma unroll
        for (int e = 0; e < E_NUM; ++e) p[e] *= inv;

        int i0 = 0;                                    // top-2, lowest-index ties
#pragma unroll
        for (int e = 1; e < E_NUM; ++e) if (p[e] > p[i0]) i0 = e;
        int i1 = (i0 == 0) ? 1 : 0;
#pragma unroll
        for (int e = 0; e < E_NUM; ++e) if (e != i0 && p[e] > p[i1]) i1 = e;

        float d  = 1.0f / (p[i0] + p[i1]);
        int pos0 = atomicAdd(&counts[i0], 1);
        tok_list[i0 * TOKENS + pos0] = token;
        w_list [i0 * TOKENS + pos0] = p[i0] * d;
        int pos1 = atomicAdd(&counts[i1], 1);
        tok_list[i1 * TOKENS + pos1] = token;
        w_list [i1 * TOKENS + pos1] = p[i1] * d;
#pragma unroll
        for (int e = 0; e < E_NUM; ++e) atomicAdd(&probsum[e], p[e]);
    }
}

// ===========================================================================
// Kernel 2: fused two-GEMM expert MLP. Block = 8 wave32s, one (expert, 16-token
// tile). X (32 KB) and H (128 KB) tiles live in the 320 KB WGP LDS; weights
// stream as pre-packed bf16 fragments: 2 global b128 + 2 ds b128 per wmma.
// ===========================================================================
__global__ __launch_bounds__(256) void moe_expert_kernel(
    const unsigned short* __restrict__ xbf,
    const unsigned short* __restrict__ w1p,
    const unsigned short* __restrict__ w2p,
    const int* __restrict__ counts,
    const int* __restrict__ tok_list, const float* __restrict__ w_list,
    float* __restrict__ out) {
    const int e    = blockIdx.x;
    const int tile = blockIdx.y;
    const int cnt  = counts[e];
    if (tile * TILE_M >= cnt) return;
    const int valid = min(TILE_M, cnt - tile * TILE_M);

    __shared__ __align__(16) unsigned short Xs[TILE_M][C_DIM + 8];
    __shared__ __align__(16) unsigned short Hs[TILE_M][F_DIM + 8];
    __shared__ int   ts[TILE_M];
    __shared__ float wr[TILE_M];

    const int tid = threadIdx.x;
    if (tid < TILE_M) {
        int idx = e * TOKENS + tile * TILE_M + tid;
        ts[tid] = (tid < valid) ? tok_list[idx] : -1;
        wr[tid] = (tid < valid) ? w_list[idx]  : 0.0f;
    }
    __syncthreads();

    // Stage X tile: pure b128 copy of pre-converted bf16 rows
    for (int i = tid; i < TILE_M * (C_DIM / 8); i += 256) {
        int r  = i >> 7;              // C_DIM/8 = 128 chunks per row
        int c8 = i & 127;
        int t  = ts[r];
        us8 v = {};
        if (t >= 0) v = ((const us8*)(xbf + (long)t * C_DIM))[c8];
        *(us8*)&Xs[r][c8 * 8] = v;
    }
    __syncthreads();

    const unsigned short* W1p = w1p + (long)e * C_DIM * F_DIM;
    const unsigned short* W2p = w2p + (long)e * F_DIM * C_DIM;
    const int wave = tid >> 5;
    const int lane = tid & 31;
    const int n    = lane & 15;
    const int mh   = (lane >> 4) << 3;   // D rows: m = mh + r (ISA 7.12.2)

    // ---- Phase 1: H = gelu(X @ W1_e); each wave owns 512 F-columns ----
    for (int sub = 0; sub < 32; ++sub) {
        const int f0 = wave * 512 + sub * 16;
        const us8* bp = (const us8*)(W1p + ((long)(f0 >> 4) * (C_DIM >> 5)) * 512)
                        + lane * 2;
        v8f acc = {};
#pragma unroll 4
        for (int kt = 0; kt < (C_DIM >> 5); ++kt) {
            __builtin_prefetch(bp + 128, 0, 1);          // 2 tiles ahead
            v16bf a = load_a_frag(&Xs[0][0], C_DIM + 8, kt << 5);
            v16bf b = concat8(bp[0], bp[1]);
            bp += 64;                                    // next 512-short tile
            acc = __builtin_amdgcn_wmma_f32_16x16x32_bf16(
                false, a, false, b, (short)0, acc, false, false);
        }
#pragma unroll
        for (int r = 0; r < 8; ++r) {
            float h = acc[r];
            float g = 0.5f * h * (1.0f + erff(h * 0.70710678118654752f)); // exact GELU
            Hs[mh + r][f0 + n] = f2bf(g);
        }
    }
    __syncthreads();

    // ---- Phase 2: Y = H @ W2_e; each wave owns 128 C-columns ----
    for (int sub = 0; sub < 8; ++sub) {
        const int c0 = wave * 128 + sub * 16;
        const us8* bp = (const us8*)(W2p + ((long)(c0 >> 4) * (F_DIM >> 5)) * 512)
                        + lane * 2;
        v8f acc = {};
#pragma unroll 4
        for (int kt = 0; kt < (F_DIM >> 5); ++kt) {
            __builtin_prefetch(bp + 128, 0, 1);
            v16bf a = load_a_frag(&Hs[0][0], F_DIM + 8, kt << 5);
            v16bf b = concat8(bp[0], bp[1]);
            bp += 64;
            acc = __builtin_amdgcn_wmma_f32_16x16x32_bf16(
                false, a, false, b, (short)0, acc, false, false);
        }
#pragma unroll
        for (int r = 0; r < 8; ++r) {
            int m = mh + r;
            int t = ts[m];
            if (t >= 0)
                atomicAdd(&out[(long)t * C_DIM + c0 + n], wr[m] * acc[r]);
        }
    }
}

// ===========================================================================
// Kernel 3: aux loss = E * sum_e f_e * P_e
// ===========================================================================
__global__ void moe_aux_kernel(const int* __restrict__ counts,
                               const float* __restrict__ probsum,
                               float* __restrict__ aux) {
    if (threadIdx.x == 0 && blockIdx.x == 0) {
        float s = 0.0f;
#pragma unroll
        for (int e = 0; e < E_NUM; ++e) {
            float f = (float)counts[e] / (float)(TOKENS * TOPK);
            float P = probsum[e] / (float)TOKENS;
            s += f * P;
        }
        *aux = (float)E_NUM * s;
    }
}

// ===========================================================================
extern "C" void kernel_launch(void* const* d_in, const int* in_sizes, int n_in,
                              void* d_out, int out_size, void* d_ws, size_t ws_size,
                              hipStream_t stream) {
    const float* x  = (const float*)d_in[0];
    const float* Wg = (const float*)d_in[1];
    const float* W1 = (const float*)d_in[2];
    const float* W2 = (const float*)d_in[3];
    float* out = (float*)d_out;

    char* ws = (char*)d_ws;
    int*            counts   = (int*)(ws + WS_COUNTS);
    float*          probsum  = (float*)(ws + WS_PROBSUM);
    int*            tok_list = (int*)(ws + WS_TOK);
    float*          w_list   = (float*)(ws + WS_WL);
    unsigned short* xbf      = (unsigned short*)(ws + WS_XBF);
    unsigned short* w1pack   = (unsigned short*)(ws + WS_W1P);
    unsigned short* w2pack   = (unsigned short*)(ws + WS_W2P);

    const int n_out = OUT_ELEMS + 1;  // final_output ++ aux_loss
    moe_zero_kernel<<<(n_out + 255) / 256, 256, 0, stream>>>(out, n_out, counts, probsum);

    const long wtot = (long)E_NUM * C_DIM * F_DIM;          // 33.5M elements each
    moe_pack_w_kernel<C_DIM, F_DIM><<<(unsigned)(wtot / 256), 256, 0, stream>>>(W1, w1pack);
    moe_pack_w_kernel<F_DIM, C_DIM><<<(unsigned)(wtot / 256), 256, 0, stream>>>(W2, w2pack);
    moe_pack_x_kernel<<<(TOKENS * C_DIM) / 256, 256, 0, stream>>>(x, xbf);

    moe_router_kernel<<<TOKENS / 8, 256, 0, stream>>>(x, Wg, counts, probsum,
                                                      tok_list, w_list);
    moe_expert_kernel<<<dim3(E_NUM, TOKENS / TILE_M), 256, 0, stream>>>(
        xbf, w1pack, w2pack, counts, tok_list, w_list, out);
    moe_aux_kernel<<<1, 32, 0, stream>>>(counts, probsum, out + OUT_ELEMS);
}
